// future_decoder_8366596293116
// MI455X (gfx1250) — matrix operand order repaired and checked by reading, not compiled, not run on hardware
//
#include <hip/hip_runtime.h>

typedef __attribute__((ext_vector_type(16))) _Float16 v16h;
typedef __attribute__((ext_vector_type(8)))  float    v8f;

#define HID       128
#define GATES4    (4*HID)      // 512
#define FUT       12
#define WAVES     8
#define AGW       16           // agents per wave
#define AGB       (WAVES*AGW)  // 128 agents per block
#define HPAD      132          // padded LDS row stride (floats) to avoid bank conflicts
#define TABH      (32*4*32*16) // 65536 f16 elements = 128 KB swizzled W_hh table

__device__ __forceinline__ float sigm_f(float x) {
    return 1.0f / (1.0f + __expf(-x));
}
__device__ __forceinline__ float tanh_f(float x) {
    return 2.0f / (1.0f + __expf(-2.0f * x)) - 1.0f;
}

// ---------------------------------------------------------------------------
// Prep: swizzle W_hh (512x128 f32, row-major) into f16 B-fragments for
// v_wmma_f32_16x16x32_f16.  B is KxN = 32x16 per fragment:
//   lanes 0-15  hold K = kf*32 + e      (e = 0..15), N = lane&15
//   lanes 16-31 hold K = kf*32 + 16 + e (e = 0..15), N = lane&15
// Table index: ((ntile*4 + kf)*32 + lane)*16 + e,  ntile = row-block of W_hh.
// ---------------------------------------------------------------------------
__global__ void prep_whh_frags(const float* __restrict__ Whh,
                               _Float16* __restrict__ frag) {
    int gid = blockIdx.x * 256 + threadIdx.x;      // 65536 total
    if (gid >= TABH) return;
    int e    = gid & 15;
    int lane = (gid >> 4) & 31;
    int kf   = (gid >> 9) & 3;
    int nt   = gid >> 11;
    int n = nt * 16 + (lane & 15);                 // row of W_hh (gate unit)
    int k = kf * 32 + ((lane < 16) ? 0 : 16) + e;  // col of W_hh (hidden in)
    frag[gid] = (_Float16)Whh[n * HID + k];
}

// ---------------------------------------------------------------------------
// Main rollout: each wave32 owns 16 agents for all 12 steps.
// h in LDS (f32 padded rows), c in registers, x in LDS, W_hh fragments in LDS.
// ---------------------------------------------------------------------------
__global__ void __launch_bounds__(256, 1)
lstm_rollout(const float* __restrict__ obs,      // (8, N, 2)
             const float* __restrict__ h0,       // (N, 128)
             const float* __restrict__ Wih,      // (512, 2)
             const float* __restrict__ bih,      // (512)
             const float* __restrict__ bhh,      // (512)
             const float* __restrict__ Wout,     // (2, 128)
             const float* __restrict__ bout,     // (2)
             const _Float16* __restrict__ bfrag, // swizzled W_hh table (global)
             float* __restrict__ out,            // (12, N, 2)
             int N) {
    __shared__ __attribute__((aligned(32))) _Float16 tab_s[TABH];  // 131072 B
    __shared__ float h_buf[WAVES][AGW * HPAD];                     //  67584 B
    __shared__ float x_buf[WAVES][AGW][2];
    __shared__ float wout_s[2 * HID];
    __shared__ float bout_s[2];
    __shared__ float bias_s[GATES4];
    __shared__ float wih0_s[GATES4];
    __shared__ float wih1_s[GATES4];

    const int tid  = threadIdx.x;
    const int wave = tid >> 5;
    const int lane = tid & 31;
    const long wa0 = (long)blockIdx.x * AGB + wave * AGW;   // first agent of wave

    // ---- async copy of the 128 KB W_hh fragment table into LDS ----
    // global_load_async_to_lds_b128: VDST = LDS byte address VGPR,
    // VADDR = 32-bit offset VGPR, SADDR = 64-bit base SGPR pair (GVS mode).
    {
        unsigned lbase = (unsigned)(unsigned long long)(uintptr_t)(void*)tab_s;
        #pragma unroll 4
        for (int i = 0; i < 32; ++i) {
            unsigned off   = (unsigned)(tid + 256 * i) * 16u;
            unsigned laddr = lbase + off;
            asm volatile("global_load_async_to_lds_b128 %0, %1, %2"
                         :: "v"(laddr), "v"(off), "s"(bfrag)
                         : "memory");
        }
        asm volatile("s_wait_asynccnt 0x0" ::: "memory");
    }

    // ---- load initial hidden state into LDS (coalesced float4) ----
    #pragma unroll 4
    for (int a = 0; a < AGW; ++a) {
        float4 v = *(const float4*)&h0[(wa0 + a) * HID + lane * 4];
        *(float4*)&h_buf[wave][a * HPAD + lane * 4] = v;
    }
    // ---- x0 = obs_traj_rel[OBS_LEN-1] ----
    {
        long base = (7L * N + wa0) * 2;
        ((float*)x_buf[wave])[lane] = obs[base + lane];
    }
    // ---- block-wide weight preload ----
    for (int i = tid; i < GATES4; i += 256) {
        bias_s[i] = bih[i] + bhh[i];
        wih0_s[i] = Wih[i * 2 + 0];
        wih1_s[i] = Wih[i * 2 + 1];
    }
    if (tid < 2 * HID) wout_s[tid] = Wout[tid];
    if (tid < 2)       bout_s[tid] = bout[tid];
    __syncthreads();

    // cell state in registers: creg[jtile][m-slot]
    float creg[8][8];
    #pragma unroll
    for (int jt = 0; jt < 8; ++jt)
        #pragma unroll
        for (int r = 0; r < 8; ++r) creg[jt][r] = 0.0f;

    const int mrow = lane & 15;               // A-layout / acc column id
    const int aoff = (lane < 16) ? 0 : 8;     // 2*aoff = A K-offset, and acc M half

    unsigned tz = 0;                          // laundered index: defeats LICM
    #pragma unroll 1
    for (int t = 0; t < FUT; ++t) {
        asm volatile("" : "+v"(tz));          // redefine each iter -> no hoist
        const v16h* tab = ((const v16h*)tab_s) + tz;

        // ---- gather x for this wave's agents into registers ----
        float xr0[8], xr1[8];
        #pragma unroll
        for (int r = 0; r < 8; ++r) {
            int m = r + aoff;
            xr0[r] = x_buf[wave][m][0];
            xr1[r] = x_buf[wave][m][1];
        }

        // ---- build f16 A-fragments (16 agents x K=128) from LDS h ----
        v16h afrag[4];
        #pragma unroll
        for (int kf = 0; kf < 4; ++kf) {
            const float* hb = &h_buf[wave][mrow * HPAD + kf * 32 + 2 * aoff];
            float4 p0 = *(const float4*)&hb[0];
            float4 p1 = *(const float4*)&hb[4];
            float4 p2 = *(const float4*)&hb[8];
            float4 p3 = *(const float4*)&hb[12];
            v16h a;
            a[0]  = (_Float16)p0.x; a[1]  = (_Float16)p0.y;
            a[2]  = (_Float16)p0.z; a[3]  = (_Float16)p0.w;
            a[4]  = (_Float16)p1.x; a[5]  = (_Float16)p1.y;
            a[6]  = (_Float16)p1.z; a[7]  = (_Float16)p1.w;
            a[8]  = (_Float16)p2.x; a[9]  = (_Float16)p2.y;
            a[10] = (_Float16)p2.z; a[11] = (_Float16)p2.w;
            a[12] = (_Float16)p3.x; a[13] = (_Float16)p3.y;
            a[14] = (_Float16)p3.z; a[15] = (_Float16)p3.w;
            afrag[kf] = a;
        }

        // ---- gates GEMM + elementwise, 16 hidden units (one j-tile) at a time
        #pragma unroll
        for (int jt = 0; jt < 8; ++jt) {
            v8f acc[4];
            // init with bias + W_ih @ x  (x is only 2-wide -> VALU)
            #pragma unroll
            for (int g = 0; g < 4; ++g) {
                int j = g * HID + jt * 16 + mrow;
                float bb = bias_s[j], w0 = wih0_s[j], w1 = wih1_s[j];
                #pragma unroll
                for (int r = 0; r < 8; ++r)
                    acc[g][r] = fmaf(xr1[r], w1, fmaf(xr0[r], w0, bb));
            }
            // h @ W_hh^T : 4 gates x 4 K-fragments of WMMA, B from LDS
            #pragma unroll
            for (int g = 0; g < 4; ++g) {
                int nt = g * 8 + jt;
                #pragma unroll
                for (int kf = 0; kf < 4; ++kf) {
                    v16h b = tab[(nt * 4 + kf) * 32 + lane];
                    acc[g] = __builtin_amdgcn_wmma_f32_16x16x32_f16(
                        false, afrag[kf], false, b,
                        (short)0, acc[g], false, false);
                }
            }
            // LSTM cell update for these 16 units (acc: i,f,g,o)
            #pragma unroll
            for (int r = 0; r < 8; ++r) {
                float cn = sigm_f(acc[1][r]) * creg[jt][r]
                         + sigm_f(acc[0][r]) * tanh_f(acc[2][r]);
                float hn = sigm_f(acc[3][r]) * tanh_f(cn);
                creg[jt][r] = cn;
                h_buf[wave][(r + aoff) * HPAD + jt * 16 + mrow] = hn;
            }
        }
        __syncthreads();

        // ---- output projection: 32 lanes = 16 agents x 2 coords ----
        {
            int a_loc = lane >> 1, coord = lane & 1;
            const float* hrow = &h_buf[wave][a_loc * HPAD];
            const float* wrow = &wout_s[coord * HID];
            float dot = bout_s[coord];
            #pragma unroll
            for (int j = 0; j < HID; j += 4) {
                float4 hv = *(const float4*)&hrow[j];
                float4 wv = *(const float4*)&wrow[j];
                dot = fmaf(hv.x, wv.x, dot);
                dot = fmaf(hv.y, wv.y, dot);
                dot = fmaf(hv.z, wv.z, dot);
                dot = fmaf(hv.w, wv.w, dot);
            }
            long agent = wa0 + a_loc;
            out[((long)t * N + agent) * 2 + coord] = dot;
            x_buf[wave][a_loc][coord] = dot;       // feeds next step's input
        }
        __syncthreads();
    }
}

extern "C" void kernel_launch(void* const* d_in, const int* in_sizes, int n_in,
                              void* d_out, int out_size, void* d_ws, size_t ws_size,
                              hipStream_t stream) {
    const float* obs  = (const float*)d_in[0];   // obs_traj_rel (8,N,2)
    // d_in[1] = fut_traj_rel (unused by reference computation)
    const float* h0   = (const float*)d_in[2];   // (N,128)
    const float* Wih  = (const float*)d_in[3];   // (512,2)
    const float* Whh  = (const float*)d_in[4];   // (512,128)
    const float* bih  = (const float*)d_in[5];
    const float* bhh  = (const float*)d_in[6];
    const float* Wout = (const float*)d_in[7];   // (2,128)
    const float* bout = (const float*)d_in[8];
    float* out = (float*)d_out;

    int N = in_sizes[2] / HID;                   // 131072
    _Float16* frag = (_Float16*)d_ws;            // 128 KB swizzled W_hh table

    prep_whh_frags<<<TABH / 256, 256, 0, stream>>>(Whh, frag);
    lstm_rollout<<<N / AGB, 256, 0, stream>>>(obs, h0, Wih, bih, bhh,
                                              Wout, bout, frag, out, N);
}
